// Decoder_2379411882257
// MI455X (gfx1250) — compile-verified, matrix operand order checked by reading, not run on hardware
//
#include <hip/hip_runtime.h>
#include <hip/hip_bf16.h>
#include <math.h>

// ---------------- sizes ----------------
#define BSZ   64
#define HH    32
#define WW    32
#define CC    684
#define VV    128
#define NPQ   512     // NP
#define QQ    256
#define EE    256
#define HIDD  256
#define LL    1024    // H*W
#define MM    65536   // BSZ*LL
#define KF    704     // C padded to mult of 32
#define KP    128     // 121 conv taps padded

// LDS strides (halfs), padded for bank-conflict-free fragment reads
#define ALDS  712     // 356 words, 356 % 64 = 36 -> distinct bank quads across 16 lanes
#define PLDS  136     // 68 words,  68 % 64 = 4  -> distinct bank quads
#define MROWS 32      // M rows staged per block (2 waves x 16)
#define SMEM_BYTES ((MROWS * ALDS + MROWS * PLDS) * 2)

typedef __attribute__((ext_vector_type(16))) __bf16 v16bf;
typedef __attribute__((ext_vector_type(8)))  float  v8f;

union V16U { v16bf v; uint4 u[2]; };

__device__ __forceinline__ unsigned short f2bf(float f) {
    unsigned int u = __float_as_uint(f);
    u += 0x7FFFu + ((u >> 16) & 1u);          // round-to-nearest-even
    return (unsigned short)(u >> 16);
}
__device__ __forceinline__ float bf2f(unsigned short s) {
    return __uint_as_float(((unsigned int)s) << 16);
}
__device__ __forceinline__ float sigmoidf_(float x) { return 1.0f / (1.0f + expf(-x)); }

// load 16 bf16 fragment elements for one lane: two contiguous 16B chunks
// (K layout of 16-bit WMMA operands: elems 0..7 at k0+8g, elems 8..15 at k0+16+8g)
__device__ __forceinline__ v16bf ld16(const unsigned short* p, int off) {
    V16U t;
    t.u[0] = *(const uint4*)(p + off);
    t.u[1] = *(const uint4*)(p + off + 16);
    return t.v;
}

// ---------------- 1) feature (b,c,l) f32 -> featbf [b*l][KF] bf16 (LDS transpose) ----
__global__ __launch_bounds__(256) void k_feat_tr(const float* __restrict__ feat,
                                                 unsigned short* __restrict__ featbf) {
    __shared__ float tile[32][33];
    const int b  = blockIdx.z;
    const int l0 = blockIdx.x * 32;
    const int c0 = blockIdx.y * 32;
    const int tx = threadIdx.x, ty = threadIdx.y;       // 32 x 8
    #pragma unroll
    for (int i = 0; i < 32; i += 8) {
        int c = c0 + ty + i, l = l0 + tx;
        float v = (c < CC) ? feat[((size_t)b * CC + c) * LL + l] : 0.0f;
        tile[ty + i][tx] = v;
    }
    __syncthreads();
    #pragma unroll
    for (int i = 0; i < 32; i += 8) {
        int l = l0 + ty + i, k = c0 + tx;
        featbf[((size_t)b * LL + l) * KF + k] = f2bf(tile[tx][ty + i]);
    }
}

// ---------------- 2) Ua_w (512,684) f32 -> UaT [512][KF] bf16 ----------------
__global__ __launch_bounds__(256) void k_ua_tr(const float* __restrict__ Ua_w,
                                               unsigned short* __restrict__ UaT) {
    int idx = blockIdx.x * 256 + threadIdx.x;
    if (idx >= NPQ * KF) return;
    int p = idx / KF, k = idx % KF;
    float v = (k < CC) ? Ua_w[p * CC + k] : 0.0f;
    UaT[idx] = f2bf(v);
}

// ---------------- 3) merged conv*Uf weights: W2t[p][128] bf16, b2[p] ----------------
__global__ __launch_bounds__(128) void k_w2(const float* __restrict__ Uf_w,
                                            const float* __restrict__ Uf_b,
                                            const float* __restrict__ convQ_w,
                                            const float* __restrict__ convQ_b,
                                            const float* __restrict__ Ua_b,
                                            unsigned short* __restrict__ W2t,
                                            float* __restrict__ b2) {
    int p = blockIdx.x, t = threadIdx.x;
    float acc = 0.0f;
    if (t < 121) {
        for (int q = 0; q < QQ; ++q)
            acc += Uf_w[p * QQ + q] * convQ_w[q * 121 + t];
    }
    W2t[p * KP + t] = f2bf(acc);
    if (t == 0) {
        float bb = Uf_b[p] + Ua_b[p];
        for (int q = 0; q < QQ; ++q) bb += convQ_b[q] * Uf_w[p * QQ + q];
        b2[p] = bb;
    }
}

// ---------------- 4) im2col patches of beta (=alpha) -> bf16 [m][128] ----------------
__global__ __launch_bounds__(128) void k_patches(const float* __restrict__ alpha_in,
                                                 unsigned short* __restrict__ patches) {
    int m = blockIdx.x, t = threadIdx.x;
    int b = m >> 10, l = m & 1023, y = l >> 5, x = l & 31;
    float v = 0.0f;
    if (t < 121) {
        int ky = t / 11, kx = t % 11;
        int yy = y + ky - 5, xx = x + kx - 5;
        if (yy >= 0 && yy < HH && xx >= 0 && xx < WW)
            v = alpha_in[b * LL + yy * WW + xx];
    }
    patches[(size_t)m * KP + t] = f2bf(v);
}

// ---------------- 5) embedding gather ----------------
__global__ __launch_bounds__(256) void k_embed(const int* __restrict__ x,
                                               const float* __restrict__ emb,
                                               float* __restrict__ embedded) {
    int b = blockIdx.x, j = threadIdx.x;
    embedded[b * EE + j] = emb[x[b] * EE + j];
}

// ---------------- 6) GRU1 ----------------
__global__ __launch_bounds__(256) void k_gru1(const float* __restrict__ embedded,
                                              const float* __restrict__ hidden,
                                              const float* __restrict__ wih,
                                              const float* __restrict__ whh,
                                              const float* __restrict__ bih,
                                              const float* __restrict__ bhh,
                                              float* __restrict__ pred) {
    int b = blockIdx.x, j = threadIdx.x;
    float gir = bih[j], giz = bih[HIDD + j], gin = bih[2 * HIDD + j];
    float ghr = bhh[j], ghz = bhh[HIDD + j], ghn = bhh[2 * HIDD + j];
    const float* e = embedded + b * EE;
    const float* h = hidden + b * HIDD;
    for (int k = 0; k < EE; ++k) {
        float ek = e[k];
        gir += ek * wih[j * EE + k];
        giz += ek * wih[(HIDD + j) * EE + k];
        gin += ek * wih[(2 * HIDD + j) * EE + k];
        float hk = h[k];
        ghr += hk * whh[j * HIDD + k];
        ghz += hk * whh[(HIDD + j) * HIDD + k];
        ghn += hk * whh[(2 * HIDD + j) * HIDD + k];
    }
    float r = sigmoidf_(gir + ghr);
    float z = sigmoidf_(giz + ghz);
    float n = tanhf(gin + r * ghn);
    pred[b * HIDD + j] = (1.0f - z) * n + z * h[j];
}

// ---------------- 7) query = pred @ Wa^T + Wa_b ----------------
__global__ __launch_bounds__(256) void k_query(const float* __restrict__ pred,
                                               const float* __restrict__ Wa_w,
                                               const float* __restrict__ Wa_b,
                                               float* __restrict__ query) {
    int idx = blockIdx.x * 256 + threadIdx.x;
    int b = idx >> 9, p = idx & 511;
    float acc = Wa_b[p];
    for (int k = 0; k < HIDD; ++k) acc += pred[b * HIDD + k] * Wa_w[p * HIDD + k];
    query[b * NPQ + p] = acc;
}

// ---------------- 8) fused WMMA attention: e_t = Va . tanh(feat@Ua + patch@W2 + bias + q)
// 2 waves/block; A (feature rows + patches) staged once in LDS, reused across all
// 8 N-tiles per group x 4 groups. B streams from global (L2/WGP$-resident, 850KB).
__global__ __launch_bounds__(64) void k_attn(const unsigned short* __restrict__ featbf,
                                             const unsigned short* __restrict__ UaT,
                                             const unsigned short* __restrict__ patches,
                                             const unsigned short* __restrict__ W2t,
                                             const float* __restrict__ b2,
                                             const float* __restrict__ query,
                                             const float* __restrict__ Va_w,
                                             const float* __restrict__ Va_b,
                                             float* __restrict__ e_t) {
    extern __shared__ unsigned short smem[];   // [MROWS][ALDS] A  + [MROWS][PLDS] patches
    const int tid  = threadIdx.x;
    const int lane = tid & 31;
    const int wave = tid >> 5;
    const int l15  = lane & 15;
    const int half = lane >> 4;
    const int mblock = blockIdx.x * MROWS;
    const int b = mblock >> 10;

    // ---- cooperative stage: 32 rows x 704 halfs of A, 32 x 128 of patches ----
    {
        const int NA = MROWS * (KF / 8);              // uint4 chunks of A
        for (int i = tid; i < NA; i += 64) {
            int row = i / (KF / 8);
            int k8  = i - row * (KF / 8);
            uint4 v = *(const uint4*)(featbf + (size_t)(mblock + row) * KF + k8 * 8);
            *(uint4*)(smem + row * ALDS + k8 * 8) = v;
        }
        unsigned short* ps = smem + MROWS * ALDS;
        const int NPch = MROWS * (KP / 8);
        for (int i = tid; i < NPch; i += 64) {
            int row = i / (KP / 8);
            int k8  = i - row * (KP / 8);
            uint4 v = *(const uint4*)(patches + (size_t)(mblock + row) * KP + k8 * 8);
            *(uint4*)(ps + row * PLDS + k8 * 8) = v;
        }
    }
    __syncthreads();

    const unsigned short* arow = smem + (wave * 16 + l15) * ALDS;               // LDS
    const unsigned short* prow = smem + MROWS * ALDS + (wave * 16 + l15) * PLDS; // LDS
    const int mbase = mblock + wave * 16;
    const float* qrow = query + b * NPQ;
    const int koff = 8 * half;

    float S[8];
    #pragma unroll
    for (int j = 0; j < 8; ++j) S[j] = 0.0f;

    for (int g = 0; g < 4; ++g) {                    // 4 groups x 8 N-tiles = N=512
        const int n0 = g * 128;
        v8f acc[8];
        #pragma unroll
        for (int t = 0; t < 8; ++t) acc[t] = (v8f){0,0,0,0,0,0,0,0};

        // feature @ UaT  (K = 704): A from LDS, B from global
        for (int kc = 0; kc < KF / 32; ++kc) {
            const int ko = kc * 32 + koff;
            v16bf a = ld16(arow, ko);
            #pragma unroll
            for (int t = 0; t < 8; ++t) {
                const unsigned short* br = UaT + (size_t)(n0 + t * 16 + l15) * KF;
                v16bf bm = ld16(br, ko);
                acc[t] = __builtin_amdgcn_wmma_f32_16x16x32_bf16(
                    false, a, false, bm, (short)0, acc[t], false, false);
            }
        }
        // patches @ W2t  (K = 128)
        #pragma unroll
        for (int kc = 0; kc < KP / 32; ++kc) {
            const int ko = kc * 32 + koff;
            v16bf a = ld16(prow, ko);
            #pragma unroll
            for (int t = 0; t < 8; ++t) {
                const unsigned short* br = W2t + (size_t)(n0 + t * 16 + l15) * KP;
                v16bf bm = ld16(br, ko);
                acc[t] = __builtin_amdgcn_wmma_f32_16x16x32_bf16(
                    false, a, false, bm, (short)0, acc[t], false, false);
            }
        }
        // epilogue: tanh + Va reduction, fused (score tensor never materialized)
        #pragma unroll
        for (int t = 0; t < 8; ++t) {
            const int n = n0 + t * 16 + l15;
            const float bias = b2[n] + qrow[n];
            const float va = Va_w[n];
            #pragma unroll
            for (int j = 0; j < 8; ++j) {
                float sc = tanhf(acc[t][j] + bias);
                S[j] += va * sc;                 // row = j + 8*half (per lane-half)
            }
        }
    }
    // reduce each row across its 16-lane half (xor offsets < 16 stay intra-half)
    #pragma unroll
    for (int j = 0; j < 8; ++j) {
        float v = S[j];
        v += __shfl_xor(v, 1, 32);
        v += __shfl_xor(v, 2, 32);
        v += __shfl_xor(v, 4, 32);
        v += __shfl_xor(v, 8, 32);
        if (l15 == 0) e_t[mbase + j + 8 * half] = v + Va_b[0];
    }
}

// ---------------- 9) softmax (exp w/o max-sub, matching reference) ----------------
__global__ __launch_bounds__(256) void k_softmax(const float* __restrict__ e_t,
                                                 float* __restrict__ alpha_out) {
    __shared__ float red[256];
    int b = blockIdx.x, tid = threadIdx.x;
    float vals[4], s = 0.0f;
    #pragma unroll
    for (int i = 0; i < 4; ++i) {
        float v = expf(e_t[b * LL + tid + i * 256]);
        vals[i] = v; s += v;
    }
    red[tid] = s; __syncthreads();
    for (int st = 128; st > 0; st >>= 1) {
        if (tid < st) red[tid] += red[tid + st];
        __syncthreads();
    }
    float inv = 1.0f / (red[0] + 1e-8f);
    #pragma unroll
    for (int i = 0; i < 4; ++i) alpha_out[b * LL + tid + i * 256] = vals[i] * inv;
}

// ---------------- 10) context[b][c] = sum_l alpha[b][l] * feat[b][l][c] (bf16 copy) ---
__global__ __launch_bounds__(256) void k_context(const unsigned short* __restrict__ featbf,
                                                 const float* __restrict__ alpha_out,
                                                 float* __restrict__ context) {
    int b = blockIdx.x, c = blockIdx.y * 256 + threadIdx.x;
    if (c >= CC) return;
    const unsigned short* fb = featbf + (size_t)b * LL * KF + c;
    const float* al = alpha_out + b * LL;
    float acc = 0.0f;
    for (int l = 0; l < LL; ++l) acc += al[l] * bf2f(fb[(size_t)l * KF]);
    context[b * CC + c] = acc;
}

// ---------------- 11) GRU2 (x = context, h = pred) -> h2 (output region) ----------------
__global__ __launch_bounds__(256) void k_gru2(const float* __restrict__ context,
                                              const float* __restrict__ pred,
                                              const float* __restrict__ wih,
                                              const float* __restrict__ whh,
                                              const float* __restrict__ bih,
                                              const float* __restrict__ bhh,
                                              float* __restrict__ h2out) {
    int b = blockIdx.x, j = threadIdx.x;
    float gir = bih[j], giz = bih[HIDD + j], gin = bih[2 * HIDD + j];
    float ghr = bhh[j], ghz = bhh[HIDD + j], ghn = bhh[2 * HIDD + j];
    const float* xv = context + b * CC;
    const float* h  = pred + b * HIDD;
    for (int k = 0; k < CC; ++k) {
        float xk = xv[k];
        gir += xk * wih[j * CC + k];
        giz += xk * wih[(HIDD + j) * CC + k];
        gin += xk * wih[(2 * HIDD + j) * CC + k];
    }
    for (int k = 0; k < HIDD; ++k) {
        float hk = h[k];
        ghr += hk * whh[j * HIDD + k];
        ghz += hk * whh[(HIDD + j) * HIDD + k];
        ghn += hk * whh[(2 * HIDD + j) * HIDD + k];
    }
    float r = sigmoidf_(gir + ghr);
    float z = sigmoidf_(giz + ghz);
    float n = tanhf(gin + r * ghn);
    h2out[b * HIDD + j] = (1.0f - z) * n + z * h[j];
}

// ---------------- 12) o = embedded + h2@Ws^T + context@Wc^T (+biases) ----------------
__global__ __launch_bounds__(256) void k_out(const float* __restrict__ embedded,
                                             const float* __restrict__ h2,
                                             const float* __restrict__ context,
                                             const float* __restrict__ Ws_w,
                                             const float* __restrict__ Ws_b,
                                             const float* __restrict__ Wc_w,
                                             const float* __restrict__ Wc_b,
                                             float* __restrict__ o_ws) {
    int b = blockIdx.x, j = threadIdx.x;
    float acc = embedded[b * EE + j] + Ws_b[j] + Wc_b[j];
    const float* hr = h2 + b * HIDD;
    for (int k = 0; k < HIDD; ++k) acc += hr[k] * Ws_w[j * HIDD + k];
    const float* cr = context + b * CC;
    for (int k = 0; k < CC; ++k) acc += cr[k] * Wc_w[j * CC + k];
    o_ws[b * EE + j] = acc;
}

// ---------------- 13) pairwise max + logits ----------------
__global__ __launch_bounds__(256) void k_logits(const float* __restrict__ o_ws,
                                                const float* __restrict__ Wo_w,
                                                const float* __restrict__ Wo_b,
                                                float* __restrict__ logits) {
    int idx = blockIdx.x * 256 + threadIdx.x;
    int b = idx >> 7, v = idx & 127;
    float acc = Wo_b[v];
    const float* orow = o_ws + b * EE;
    for (int e = 0; e < EE / 2; ++e) {
        float m = fmaxf(orow[2 * e], orow[2 * e + 1]);
        acc += m * Wo_w[v * (EE / 2) + e];
    }
    logits[b * VV + v] = acc;
}

// ---------------- host launcher ----------------
extern "C" void kernel_launch(void* const* d_in, const int* in_sizes, int n_in,
                              void* d_out, int out_size, void* d_ws, size_t ws_size,
                              hipStream_t stream) {
    (void)in_sizes; (void)n_in; (void)out_size; (void)ws_size;
    const int*   x        = (const int*)  d_in[0];
    const float* hidden   = (const float*)d_in[1];
    const float* feature  = (const float*)d_in[2];
    const float* alpha_in = (const float*)d_in[3];
    const float* emb      = (const float*)d_in[4];
    const float* g1_wih   = (const float*)d_in[5];
    const float* g1_whh   = (const float*)d_in[6];
    const float* g1_bih   = (const float*)d_in[7];
    const float* g1_bhh   = (const float*)d_in[8];
    const float* g2_wih   = (const float*)d_in[9];
    const float* g2_whh   = (const float*)d_in[10];
    const float* g2_bih   = (const float*)d_in[11];
    const float* g2_bhh   = (const float*)d_in[12];
    const float* convQ_w  = (const float*)d_in[13];
    const float* convQ_b  = (const float*)d_in[14];
    const float* Wa_w     = (const float*)d_in[15];
    const float* Wa_b     = (const float*)d_in[16];
    const float* Ua_w     = (const float*)d_in[17];
    const float* Ua_b     = (const float*)d_in[18];
    const float* Uf_w     = (const float*)d_in[19];
    const float* Uf_b     = (const float*)d_in[20];
    const float* Va_w     = (const float*)d_in[21];
    const float* Va_b     = (const float*)d_in[22];
    const float* Ws_w     = (const float*)d_in[23];
    const float* Ws_b     = (const float*)d_in[24];
    const float* Wc_w     = (const float*)d_in[25];
    const float* Wc_b     = (const float*)d_in[26];
    const float* Wo_w     = (const float*)d_in[27];
    const float* Wo_b     = (const float*)d_in[28];

    float* out      = (float*)d_out;
    float* logits   = out;                 // (64,128)
    float* h2out    = out + BSZ * VV;      // (64,256)
    float* alphaout = out + BSZ * VV + BSZ * HIDD; // (64,1024)

    // workspace layout
    char* ws = (char*)d_ws;
    size_t off = 0;
    unsigned short* featbf  = (unsigned short*)(ws + off); off += (size_t)MM * KF * 2;
    unsigned short* UaT     = (unsigned short*)(ws + off); off += (size_t)NPQ * KF * 2;
    unsigned short* patches = (unsigned short*)(ws + off); off += (size_t)MM * KP * 2;
    unsigned short* W2t     = (unsigned short*)(ws + off); off += (size_t)NPQ * KP * 2;
    float* b2       = (float*)(ws + off); off += NPQ * 4;
    float* embedded = (float*)(ws + off); off += BSZ * EE * 4;
    float* pred     = (float*)(ws + off); off += BSZ * HIDD * 4;
    float* query    = (float*)(ws + off); off += BSZ * NPQ * 4;
    float* e_t      = (float*)(ws + off); off += BSZ * LL * 4;
    float* context  = (float*)(ws + off); off += BSZ * CC * 4;
    float* o_ws     = (float*)(ws + off); off += BSZ * EE * 4;

    // 1-4: data prep for the fused WMMA GEMM
    k_feat_tr<<<dim3(LL / 32, KF / 32, BSZ), dim3(32, 8), 0, stream>>>(feature, featbf);
    k_ua_tr<<<(NPQ * KF + 255) / 256, 256, 0, stream>>>(Ua_w, UaT);
    k_w2<<<NPQ, KP, 0, stream>>>(Uf_w, Uf_b, convQ_w, convQ_b, Ua_b, W2t, b2);
    k_patches<<<MM, KP, 0, stream>>>(alpha_in, patches);

    // 5-7: embedding, GRU1, query
    k_embed<<<BSZ, EE, 0, stream>>>(x, emb, embedded);
    k_gru1<<<BSZ, HIDD, 0, stream>>>(embedded, hidden, g1_wih, g1_whh, g1_bih, g1_bhh, pred);
    k_query<<<(BSZ * NPQ) / 256, 256, 0, stream>>>(pred, Wa_w, Wa_b, query);

    // 8: fused attention GEMM (WMMA bf16), A staged in LDS, 2048 blocks x 2 waves
    k_attn<<<MM / MROWS, 64, SMEM_BYTES, stream>>>(featbf, UaT, patches, W2t, b2,
                                                   query, Va_w, Va_b, e_t);

    // 9-10: softmax + context
    k_softmax<<<BSZ, 256, 0, stream>>>(e_t, alphaout);
    k_context<<<dim3(BSZ, (CC + 255) / 256), 256, 0, stream>>>(featbf, alphaout, context);

    // 11-13: GRU2, output projection, logits
    k_gru2<<<BSZ, HIDD, 0, stream>>>(context, pred, g2_wih, g2_whh, g2_bih, g2_bhh, h2out);
    k_out<<<BSZ, EE, 0, stream>>>(embedded, h2out, context, Ws_w, Ws_b, Wc_w, Wc_b, o_ws);
    k_logits<<<(BSZ * VV) / 256, 256, 0, stream>>>(o_ws, Wo_w, Wo_b, logits);
}
